// TransformerAttentionModule_28991029248861
// MI455X (gfx1250) — matrix operand order repaired and checked by reading, not compiled, not run on hardware
//
#include <hip/hip_runtime.h>
#include <math.h>

#define DIMD  128
#define HEADS 8
#define HD    16

typedef __attribute__((ext_vector_type(2)))  float  v2f;
typedef __attribute__((ext_vector_type(8)))  float  v8f;
typedef __attribute__((ext_vector_type(16))) __bf16 v16bf;

#if defined(__has_builtin)
#if __has_builtin(__builtin_amdgcn_wmma_f32_16x16x4_f32)
#define USE_F32_WMMA 1
#endif
#endif

// -------- float atomic max via order-preserving integer encoding --------
__device__ __forceinline__ void atomicMaxF(float* addr, float val) {
    if (val >= 0.0f) atomicMax((int*)addr, __float_as_int(val));
    else             atomicMin((unsigned int*)addr, __float_as_uint(val));
}

// ---------------------------------------------------------------------------
// GEMM: O[mat] = X @ W[mat] + B[mat], X is [nrows x 128], W is [128 x 128].
// Block = 256 threads = 8 wave32; block covers one 16-row strip; wave w owns
// columns [16w, 16w+16). blockIdx.y selects which (W,B,O) triple.
// Uses V_WMMA_F32_16X16X4_F32 (fp32 matrix core path) when available.
// ---------------------------------------------------------------------------
__global__ __launch_bounds__(256)
void gat_gemm16(const float* __restrict__ X, int nrows,
                const float* __restrict__ W0, const float* __restrict__ B0, float* __restrict__ O0,
                const float* __restrict__ W1, const float* __restrict__ B1, float* __restrict__ O1,
                const float* __restrict__ W2, const float* __restrict__ B2, float* __restrict__ O2)
{
    __shared__ float xs[16][132];           // padded stride: conflict-free reads

    const int mat = blockIdx.y;
    const float* __restrict__ W  = (mat == 0) ? W0 : (mat == 1) ? W1 : W2;
    const float* __restrict__ Bv = (mat == 0) ? B0 : (mat == 1) ? B1 : B2;
    float*       __restrict__ O  = (mat == 0) ? O0 : (mat == 1) ? O1 : O2;

    const int row0 = blockIdx.x * 16;
    const int t    = threadIdx.x;

    // stage 16x128 fp32 x-tile into LDS (512 float4s, 2 per thread)
    for (int f = t; f < 512; f += 256) {
        int r  = f >> 5;                    // 32 float4 per row
        int c4 = f & 31;
        int rg = row0 + r; if (rg >= nrows) rg = nrows - 1;
        float4 val = ((const float4*)(X + (size_t)rg * DIMD))[c4];
        *(float4*)&xs[r][c4 * 4] = val;
    }
    __syncthreads();

    const int wave = t >> 5;                // wave32
    const int lane = t & 31;
    const int hf   = lane >> 4;             // lane half (0/1)
    const int l    = lane & 15;
    const int col0 = wave * 16;

    v8f acc = {0.f, 0.f, 0.f, 0.f, 0.f, 0.f, 0.f, 0.f};

#if defined(USE_F32_WMMA)
    // A 16x4 fp32: VGPR j holds K = 2*half + j ; rows = lane%16
    // B 4x16 fp32: VGPR j holds row K = 2*half + j, col = lane%16
    #pragma unroll
    for (int kk = 0; kk < DIMD; kk += 4) {
        const int ka = kk + 2 * hf;
        v2f a, b;
        a[0] = xs[l][ka];
        a[1] = xs[l][ka + 1];
        b[0] = W[(size_t)ka       * DIMD + col0 + l];
        b[1] = W[(size_t)(ka + 1) * DIMD + col0 + l];
        acc = __builtin_amdgcn_wmma_f32_16x16x4_f32(
                  false, a, false, b, (short)0, acc, false, false);
    }
#else
    // Fallback: bf16 16x16x32 (codegen-confirmed builtin), fp32 accumulate.
    #pragma unroll
    for (int kk = 0; kk < DIMD; kk += 32) {
        v16bf a, b;
        #pragma unroll
        for (int i = 0; i < 8; ++i)  a[i] = (__bf16)xs[l][kk + 8 * hf + i];
        #pragma unroll
        for (int i = 8; i < 16; ++i) a[i] = (__bf16)xs[l][kk + 16 + 8 * hf + (i - 8)];
        #pragma unroll
        for (int i = 0; i < 16; ++i)
            b[i] = (__bf16)W[(size_t)(kk + 8 * hf + i + 8 * (i >> 3)) * DIMD + col0 + l];
        acc = __builtin_amdgcn_wmma_f32_16x16x32_bf16(
                  false, a, false, b, (short)0, acc, false, false);
    }
#endif

    // C/D layout: VGPR r -> row r + 8*half, col = lane%16
    const float bias = Bv[col0 + l];
    #pragma unroll
    for (int r = 0; r < 8; ++r) {
        const int row = row0 + r + 8 * hf;
        if (row < nrows) O[(size_t)row * DIMD + col0 + l] = acc[r] + bias;
    }
}

// ---------------------------------------------------------------------------
// Init: m = -inf, z = 0, agg = 0  (must run every call; ws is poisoned once)
// ---------------------------------------------------------------------------
__global__ __launch_bounds__(256)
void gat_init(float* __restrict__ m, float* __restrict__ z,
              float* __restrict__ agg, int n)
{
    size_t i  = (size_t)blockIdx.x * blockDim.x + threadIdx.x;
    size_t nh = (size_t)n * HEADS;
    if (i < (size_t)n * DIMD) agg[i] = 0.0f;
    if (i < nh) { m[i] = -__builtin_inff(); z[i] = 0.0f; }
}

// ---------------------------------------------------------------------------
// Pass 1: scores[e,h] = dot(q[src[e],h,:], k[dst[e],h,:]) / 4 ; segment max
// ---------------------------------------------------------------------------
__global__ __launch_bounds__(256)
void gat_score(const float* __restrict__ q, const float* __restrict__ k,
               const int* __restrict__ src, const int* __restrict__ dst,
               float* __restrict__ scores, float* __restrict__ m, long eh)
{
    long idx = (long)blockIdx.x * blockDim.x + threadIdx.x;
    if (idx >= eh) return;
    int e = (int)(idx >> 3);
    int h = (int)(idx & 7);
    int s = src[e], d = dst[e];
    const float4* qr = (const float4*)(q + (size_t)s * DIMD + h * HD);
    const float4* kr = (const float4*)(k + (size_t)d * DIMD + h * HD);
    float acc = 0.f;
    #pragma unroll
    for (int i = 0; i < 4; ++i) {
        float4 a = qr[i], b = kr[i];
        acc += a.x * b.x + a.y * b.y + a.z * b.z + a.w * b.w;
    }
    acc *= 0.25f;                         // 1/sqrt(HD=16)
    scores[idx] = acc;
    atomicMaxF(&m[(size_t)d * HEADS + h], acc);
}

// ---------------------------------------------------------------------------
// Pass 2: z[dst,h] += exp(score - m[dst,h])
// ---------------------------------------------------------------------------
__global__ __launch_bounds__(256)
void gat_expsum(const float* __restrict__ scores, const int* __restrict__ dst,
                const float* __restrict__ m, float* __restrict__ z, long eh)
{
    long idx = (long)blockIdx.x * blockDim.x + threadIdx.x;
    if (idx >= eh) return;
    int e = (int)(idx >> 3);
    int h = (int)(idx & 7);
    int d = dst[e];
    float ex = __expf(scores[idx] - m[(size_t)d * HEADS + h]);
    atomicAdd(&z[(size_t)d * HEADS + h], ex);
}

// ---------------------------------------------------------------------------
// Pass 3: agg[dst,h,:] += v[src,h,:] * exp(score - m)/z
// ---------------------------------------------------------------------------
__global__ __launch_bounds__(256)
void gat_scatter(const float* __restrict__ scores, const int* __restrict__ src,
                 const int* __restrict__ dst, const float* __restrict__ m,
                 const float* __restrict__ z, const float* __restrict__ v,
                 float* __restrict__ agg, long eh)
{
    long idx = (long)blockIdx.x * blockDim.x + threadIdx.x;
    if (idx >= eh) return;
    int e = (int)(idx >> 3);
    int h = (int)(idx & 7);
    int s = src[e], d = dst[e];
    float attn = __expf(scores[idx] - m[(size_t)d * HEADS + h])
                 / z[(size_t)d * HEADS + h];
    const float4* vr = (const float4*)(v + (size_t)s * DIMD + h * HD);
    float* ar = agg + (size_t)d * DIMD + h * HD;
    #pragma unroll
    for (int i = 0; i < 4; ++i) {
        float4 vv = vr[i];
        atomicAdd(ar + i * 4 + 0, vv.x * attn);
        atomicAdd(ar + i * 4 + 1, vv.y * attn);
        atomicAdd(ar + i * 4 + 2, vv.z * attn);
        atomicAdd(ar + i * 4 + 3, vv.w * attn);
    }
}

// ---------------------------------------------------------------------------
extern "C" void kernel_launch(void* const* d_in, const int* in_sizes, int n_in,
                              void* d_out, int out_size, void* d_ws, size_t ws_size,
                              hipStream_t stream)
{
    (void)n_in; (void)out_size; (void)ws_size;

    const float* x   = (const float*)d_in[0];
    const int*   src = (const int*)  d_in[1];
    const int*   dst = (const int*)  d_in[2];
    const float* Wq  = (const float*)d_in[3];
    const float* bq  = (const float*)d_in[4];
    const float* Wk  = (const float*)d_in[5];
    const float* bk  = (const float*)d_in[6];
    const float* Wv  = (const float*)d_in[7];
    const float* bv  = (const float*)d_in[8];
    const float* Wo  = (const float*)d_in[9];
    const float* bo  = (const float*)d_in[10];

    const int  n  = in_sizes[0] / DIMD;        // 100000
    const int  eN = in_sizes[1];               // 1600000
    const long eh = (long)eN * HEADS;

    // workspace layout (floats)
    float* ws     = (float*)d_ws;
    size_t nd     = (size_t)n * DIMD;
    float* q      = ws;
    float* kbuf   = q    + nd;
    float* v      = kbuf + nd;
    float* agg    = v    + nd;
    float* scores = agg  + nd;
    float* m      = scores + (size_t)eN * HEADS;
    float* z      = m      + (size_t)n  * HEADS;

    float* out = (float*)d_out;

    const int ntiles = (n + 15) / 16;

    // init m/z/agg
    {
        int blocks = (int)((nd + 255) / 256);
        gat_init<<<blocks, 256, 0, stream>>>(m, z, agg, n);
    }

    // fused QKV projection (3 WMMA GEMMs via blockIdx.y)
    gat_gemm16<<<dim3(ntiles, 3), 256, 0, stream>>>(
        x, n, Wq, bq, q, Wk, bk, kbuf, Wv, bv, v);

    // edge softmax: scores+max, exp-sum, weighted scatter
    {
        int blocks = (int)((eh + 255) / 256);
        gat_score  <<<blocks, 256, 0, stream>>>(q, kbuf, src, dst, scores, m, eh);
        gat_expsum <<<blocks, 256, 0, stream>>>(scores, dst, m, z, eh);
        gat_scatter<<<blocks, 256, 0, stream>>>(scores, src, dst, m, z, v, agg, eh);
    }

    // output projection
    gat_gemm16<<<dim3(ntiles, 1), 256, 0, stream>>>(
        agg, n, Wo, bo, out, Wo, bo, out, Wo, bo, out);
}
